// FlaxGPTNeoLocalSelfAttention_86431921864935
// MI455X (gfx1250) — matrix operand order repaired and checked by reading, not compiled, not run on hardware
//
#include <hip/hip_runtime.h>
#include <hip/hip_bf16.h>
#include <math.h>
#include <cstdint>

typedef _Float16 half_t;
typedef __attribute__((ext_vector_type(16))) _Float16 v16h;
typedef __attribute__((ext_vector_type(8)))  _Float16 v8h;
typedef __attribute__((ext_vector_type(8)))  float    v8f;

// Problem constants (fixed by the reference)
constexpr int Bn  = 2;
constexpr int Sn  = 4096;
constexpr int Dn  = 2048;
constexpr int Hn  = 16;
constexpr int HDn = 128;
constexpr int Wn  = 256;      // window
constexpr int BLn = 256;      // block length
constexpr int NBn = Sn / BLn; // 16
constexpr int Mn  = Bn * Sn;  // 8192 rows for the projection GEMMs

__device__ __forceinline__ v16h cat8(v8h lo, v8h hi) {
  return __builtin_shufflevector(lo, hi, 0,1,2,3,4,5,6,7,8,9,10,11,12,13,14,15);
}
__device__ __forceinline__ v8f wmma32(v16h a, v16h b, v8f c) {
  // D = A(16x32 f16) * B(32x16 f16) + C(16x16 f32)
  return __builtin_amdgcn_wmma_f32_16x16x32_f16(false, a, false, b, (short)0, c, false, false);
}

// ---------------------------------------------------------------------------
// CDNA5 async global -> LDS copy (ASYNCcnt path). GVS addressing:
//   mem_addr = SGPR base + signed 32-bit VGPR offset.
// vdst VGPR carries the LDS byte address (generic shared ptr truncated to 32b
// == addrspace(3) offset). 16 bytes per lane land directly in LDS, no VGPRs.
// ---------------------------------------------------------------------------
__device__ __forceinline__ void async_copy_b128(uint32_t lds_off, const void* base,
                                                uint32_t voff) {
  asm volatile("global_load_async_to_lds_b128 %0, %1, %2"
               :: "v"(lds_off), "v"(voff), "s"((uint64_t)(uintptr_t)base)
               : "memory");
}
#if __has_builtin(__builtin_amdgcn_s_wait_asynccnt)
#define WAIT_ASYNC(n) __builtin_amdgcn_s_wait_asynccnt(n)
#else
#define WAIT_ASYNC(n) asm volatile("s_wait_asynccnt %0" :: "i"(n) : "memory")
#endif
__device__ __forceinline__ uint32_t lds_addr(const void* p) {
  return (uint32_t)(uintptr_t)p;   // generic shared ptr: low 32 bits = LDS offset
}

// ---------------------------------------------------------------------------
// Elementwise f32 -> f16 cast (8 elements / thread, 16B loads & stores)
// ---------------------------------------------------------------------------
__global__ __launch_bounds__(256)
void cast_f32_to_f16_kernel(const float* __restrict__ x, half_t* __restrict__ y, int n8) {
  int i = blockIdx.x * blockDim.x + threadIdx.x;
  if (i >= n8) return;
  const float4* src = (const float4*)x;
  float4 a = src[2*i], b = src[2*i+1];
  v8h o = { (half_t)a.x,(half_t)a.y,(half_t)a.z,(half_t)a.w,
            (half_t)b.x,(half_t)b.y,(half_t)b.z,(half_t)b.w };
  *(v8h*)(y + (size_t)8*i) = o;
}

// ---------------------------------------------------------------------------
// W[k][n] f32 -> Wt[n][k] f16  (tiled LDS transpose, D x D with D=2048)
// ---------------------------------------------------------------------------
__global__ __launch_bounds__(256)
void transpose_cast_kernel(const float* __restrict__ W, half_t* __restrict__ Wt) {
  __shared__ float tile[32][33];
  int x  = blockIdx.x*32 + threadIdx.x;
  int y0 = blockIdx.y*32 + threadIdx.y;
  #pragma unroll
  for (int j = 0; j < 4; ++j)
    tile[threadIdx.y + 8*j][threadIdx.x] = W[(size_t)(y0 + 8*j)*Dn + x];
  __syncthreads();
  int xo  = blockIdx.y*32 + threadIdx.x;
  int yo0 = blockIdx.x*32 + threadIdx.y;
  #pragma unroll
  for (int j = 0; j < 4; ++j)
    Wt[(size_t)(yo0 + 8*j)*Dn + xo] = (half_t)tile[threadIdx.x][threadIdx.y + 8*j];
}

// ---------------------------------------------------------------------------
// Tiled WMMA GEMM:  C[M,N] = A[M,K](f16,row-major) * Bt[N,K](f16,row-major)^T
// Workgroup tile 128x128, 8 waves (2x4), wave tile 64x32, BK=32.
// Double-buffered LDS filled by async global->LDS DMA; stage n+1 is in flight
// while stage n feeds the WMMAs. In-order ASYNCcnt completion: after issuing
// the next stage's 4 loads, s_wait_asynccnt 4 guarantees this stage landed.
// MODE 0: f16 output with scale.  MODE 1: f32 output + bias.
// LDS rows padded to 40 f16 (80B) -> conflict-free b128 fragment gathers.
// ---------------------------------------------------------------------------
template<int MODE>
__global__ __launch_bounds__(256)
void gemm_wmma_kernel(const half_t* __restrict__ A, const half_t* __restrict__ Bt,
                      half_t* __restrict__ out16, float* __restrict__ out32,
                      const float* __restrict__ bias,
                      int M, int N, int K, float scale) {
  constexpr int LDT = 40;
  __shared__ half_t As[2][128*LDT];
  __shared__ half_t Bs[2][128*LDT];

  const int tid  = threadIdx.x;
  const int wave = tid >> 5, lane = tid & 31;
  const int wm = wave >> 2, wn = wave & 3;       // 2 x 4 wave grid
  const int hl = lane >> 4, lm = lane & 15;      // lane half, lane-in-half
  const int m0 = blockIdx.y * 128, n0 = blockIdx.x * 128;

  v8f acc[4][2] = {};

  // staging: 2 threads per LDS row, 16 f16 (2 x b128) each
  const int sr = tid >> 1, sc = (tid & 1) * 16;
  const uint32_t aoff0 = (uint32_t)(((m0 + sr)*(size_t)K + sc) * sizeof(half_t));
  const uint32_t boff0 = (uint32_t)(((n0 + sr)*(size_t)K + sc) * sizeof(half_t));
  uint32_t aLds[2], bLds[2];
  #pragma unroll
  for (int s = 0; s < 2; ++s) {
    aLds[s] = lds_addr(&As[s][sr*LDT + sc]);
    bLds[s] = lds_addr(&Bs[s][sr*LDT + sc]);
  }
  auto issue = [&](int buf, int k0) {
    const uint32_t ao = aoff0 + (uint32_t)k0 * sizeof(half_t);
    const uint32_t bo = boff0 + (uint32_t)k0 * sizeof(half_t);
    async_copy_b128(aLds[buf],      A,  ao);
    async_copy_b128(aLds[buf] + 16, A,  ao + 16);
    async_copy_b128(bLds[buf],      Bt, bo);
    async_copy_b128(bLds[buf] + 16, Bt, bo + 16);
  };

  issue(0, 0);
  for (int k0 = 0; k0 < K; k0 += 32) {
    const int cur = (k0 >> 5) & 1;
    if (k0 + 32 < K) {
      issue(cur ^ 1, k0 + 32);   // prefetch next stage (buffer freed by the
      WAIT_ASYNC(4);             // trailing barrier of the previous iteration)
    } else {
      WAIT_ASYNC(0);
    }
    __syncthreads();

    v16h af[4], bf[2];
    #pragma unroll
    for (int mf = 0; mf < 4; ++mf) {             // A frag: M=lm, K split by lane half
      int row = wm*64 + mf*16 + lm;
      af[mf] = cat8(*(const v8h*)&As[cur][row*LDT + hl*8],
                    *(const v8h*)&As[cur][row*LDT + 16 + hl*8]);
    }
    #pragma unroll
    for (int nf = 0; nf < 2; ++nf) {             // B frag: N=lm, K split by lane half
      int row = wn*32 + nf*16 + lm;
      bf[nf] = cat8(*(const v8h*)&Bs[cur][row*LDT + hl*8],
                    *(const v8h*)&Bs[cur][row*LDT + 16 + hl*8]);
    }
    #pragma unroll
    for (int mf = 0; mf < 4; ++mf)
      #pragma unroll
      for (int nf = 0; nf < 2; ++nf)
        acc[mf][nf] = wmma32(af[mf], bf[nf], acc[mf][nf]);
    __syncthreads();                             // stage buffer reusable
  }

  #pragma unroll
  for (int mf = 0; mf < 4; ++mf)
    #pragma unroll
    for (int nf = 0; nf < 2; ++nf)
      #pragma unroll
      for (int r = 0; r < 8; ++r) {              // C frag: row hl*8+r, col lm
        int gm = m0 + wm*64 + mf*16 + hl*8 + r;
        int gn = n0 + wn*32 + nf*16 + lm;
        if (MODE == 0) out16[(size_t)gm*N + gn] = (half_t)(acc[mf][nf][r] * scale);
        else           out32[(size_t)gm*N + gn] = acc[mf][nf][r] + bias[gn];
      }
}

// ---------------------------------------------------------------------------
// Local (windowed, causal) attention. One workgroup per (b, block, head).
// 16 waves x 16 query rows = 256 queries; 4 key tiles of 128 (W+BL = 512).
// Valid keys for query q (tile-local coords): q < t <= q+256, and kpos >= 0.
// K tile is staged via async global->LDS DMA (straight copy); V needs a VGPR
// round-trip for its transpose. Q is pre-scaled by 1/sqrt(HD).
// ---------------------------------------------------------------------------
__global__ __launch_bounds__(512)
void attn_kernel(const half_t* __restrict__ Qh, const half_t* __restrict__ Kh,
                 const half_t* __restrict__ Vh, half_t* __restrict__ Oh) {
  constexpr int LDK = HDn + 8;  // K tile row stride (f16), 16B-aligned rows
  constexpr int LDV = 128 + 8;  // V^T tile row stride
  constexpr int LDP = 128 + 8;  // P staging row stride
  extern __shared__ char smem_raw[];
  half_t* Ks  = (half_t*)smem_raw;       // [128 keys][LDK]  (K[key][d])
  half_t* Vts = Ks  + 128*LDK;           // [128 d][LDV]     (V^T[d][key])
  half_t* Ps  = Vts + HDn*LDV;           // [16 waves][16 q][LDP]

  const int bid = blockIdx.x;
  const int h  = bid % Hn;
  const int nb = (bid / Hn) % NBn;
  const int b  = bid / (Hn * NBn);

  const int tid  = threadIdx.x;
  const int wave = tid >> 5, lane = tid & 31;
  const int hl = lane >> 4, lm = lane & 15;
  const int q0 = wave * 16;
  const int kbase = nb*BLn - Wn;         // global key position of tile coord t=0

  // Q A-fragments kept in registers for all 4 HD k-steps (scale baked in).
  v16h qa[4];
  {
    const size_t qrow = (size_t)(b*Sn + nb*BLn + q0 + lm)*Dn + h*HDn;
    #pragma unroll
    for (int ks = 0; ks < 4; ++ks)
      qa[ks] = cat8(*(const v8h*)&Qh[qrow + ks*32 + hl*8],
                    *(const v8h*)&Qh[qrow + ks*32 + 16 + hl*8]);
  }

  v8f oacc[8] = {};
  float mrun[8], lrun[8];
  #pragma unroll
  for (int r = 0; r < 8; ++r) { mrun[r] = -1e30f; lrun[r] = 0.f; }

  half_t* Pw = Ps + wave * (16 * LDP);

  for (int kt = 0; kt < 4; ++kt) {
    const int t0 = kt * 128;
    __syncthreads();
    // -------- stage K (async DMA) and V transposed (VGPR round-trip) --------
    {
      const int key = tid >> 2, seg = tid & 3;     // 4 threads per key row
      const int kpos = kbase + t0 + key;
      half_t* ksp = &Ks[key*LDK + seg*32];
      if (kpos >= 0) {
        const size_t row = (size_t)(b*Sn + kpos)*Dn + h*HDn + seg*32;
        const uint32_t ldsK = lds_addr(ksp);
        const uint32_t voff = (uint32_t)(row * sizeof(half_t));
        #pragma unroll
        for (int i = 0; i < 4; ++i)
          async_copy_b128(ldsK + 16*i, Kh, voff + 16*i);
        v8h vv[4];
        #pragma unroll
        for (int i = 0; i < 4; ++i) vv[i] = *(const v8h*)&Vh[row + 8*i];
        #pragma unroll
        for (int i = 0; i < 4; ++i)
          #pragma unroll
          for (int j = 0; j < 8; ++j)
            Vts[(seg*32 + 8*i + j)*LDV + key] = vv[i][j];
      } else {
        v8h z = {};
        #pragma unroll
        for (int i = 0; i < 4; ++i) *(v8h*)&ksp[8*i] = z;
        #pragma unroll
        for (int i = 0; i < 4; ++i)
          #pragma unroll
          for (int j = 0; j < 8; ++j)
            Vts[(seg*32 + 8*i + j)*LDV + key] = (half_t)0;
      }
    }
    WAIT_ASYNC(0);                 // DMA landed (barrier below covers DS stores)
    __syncthreads();

    const bool active = !((t0 + 127 <= q0) ||            // all t <= q : masked
                          (t0 > q0 + 15 + Wn) ||         // all t >  q+W : masked
                          (nb == 0 && t0 + 127 < Wn));   // look-back pad region
    if (active) {
      // -------- S = Q K^T  (16 q x 128 keys per wave) --------
      v8f sacc[8] = {};
      #pragma unroll
      for (int ks = 0; ks < 4; ++ks)
        #pragma unroll
        for (int nf = 0; nf < 8; ++nf) {
          const int krow = nf*16 + lm;                  // B-frag lane N = key
          v16h bf = cat8(*(const v8h*)&Ks[krow*LDK + ks*32 + hl*8],
                         *(const v8h*)&Ks[krow*LDK + ks*32 + 16 + hl*8]);
          sacc[nf] = wmma32(qa[ks], bf, sacc[nf]);
        }
      // -------- masked online softmax; emit P (f16) to LDS --------
      #pragma unroll
      for (int r = 0; r < 8; ++r) {
        const int qg = q0 + hl*8 + r;                   // query row of c[r]
        float mt = -1e30f;
        #pragma unroll
        for (int nf = 0; nf < 8; ++nf) {
          const int t = t0 + nf*16 + lm;
          const bool valid = (t > qg) && (t <= qg + Wn) && (kbase + t >= 0);
          mt = fmaxf(mt, valid ? sacc[nf][r] : -1e30f);
        }
        #pragma unroll
        for (int off = 1; off < 16; off <<= 1)          // row spans 16 lanes
          mt = fmaxf(mt, __shfl_xor(mt, off, 32));
        const float mnew  = fmaxf(mrun[r], mt);
        const float alpha = __expf(mrun[r] - mnew);
        float rsum = 0.f;
        #pragma unroll
        for (int nf = 0; nf < 8; ++nf) {
          const int t = t0 + nf*16 + lm;
          const bool valid = (t > qg) && (t <= qg + Wn) && (kbase + t >= 0);
          const float p = valid ? __expf(sacc[nf][r] - mnew) : 0.f;
          rsum += p;
          Pw[(hl*8 + r)*LDP + nf*16 + lm] = (half_t)p;
        }
        #pragma unroll
        for (int off = 1; off < 16; off <<= 1)
          rsum += __shfl_xor(rsum, off, 32);
        lrun[r] = lrun[r]*alpha + rsum;
        mrun[r] = mnew;
        #pragma unroll
        for (int df = 0; df < 8; ++df)
          oacc[df][r] = oacc[df][r] * alpha;
      }
    }
    __syncthreads();   // P writes (C-layout) -> P reads (A-layout)
    if (active) {
      // -------- O += P V  --------
      #pragma unroll
      for (int ks = 0; ks < 4; ++ks) {
        v16h pa = cat8(*(const v8h*)&Pw[lm*LDP + ks*32 + hl*8],
                       *(const v8h*)&Pw[lm*LDP + ks*32 + 16 + hl*8]);
        #pragma unroll
        for (int df = 0; df < 8; ++df) {
          const int nd = df*16 + lm;                    // B-frag lane N = d
          v16h bf = cat8(*(const v8h*)&Vts[nd*LDV + ks*32 + hl*8],
                         *(const v8h*)&Vts[nd*LDV + ks*32 + 16 + hl*8]);
          oacc[df] = wmma32(pa, bf, oacc[df]);
        }
      }
    }
  }

  // -------- normalize and store f16 for the output projection --------
  float inv[8];
  #pragma unroll
  for (int r = 0; r < 8; ++r) inv[r] = 1.f / lrun[r];   // >0: self key always valid
  #pragma unroll
  for (int df = 0; df < 8; ++df)
    #pragma unroll
    for (int r = 0; r < 8; ++r) {
      const size_t orow = (size_t)(b*Sn + nb*BLn + q0 + hl*8 + r);
      Oh[orow*Dn + h*HDn + df*16 + lm] = (half_t)(oacc[df][r] * inv[r]);
    }
}

// ---------------------------------------------------------------------------
extern "C" void kernel_launch(void* const* d_in, const int* in_sizes, int n_in,
                              void* d_out, int out_size, void* d_ws, size_t ws_size,
                              hipStream_t stream) {
  (void)in_sizes; (void)n_in; (void)out_size; (void)ws_size;
  const float* X  = (const float*)d_in[0];
  const float* Wq = (const float*)d_in[1];
  const float* Wk = (const float*)d_in[2];
  const float* Wv = (const float*)d_in[3];
  const float* Wo = (const float*)d_in[4];
  const float* bo = (const float*)d_in[5];
  float* out = (float*)d_out;

  char* ws = (char*)d_ws;
  const size_t XH_BYTES = (size_t)Mn * Dn * sizeof(half_t);   // 33.5 MB
  const size_t W_BYTES  = (size_t)Dn * Dn * sizeof(half_t);   //  8.4 MB
  half_t* Xh  = (half_t*)ws; ws += XH_BYTES;
  half_t* WqT = (half_t*)ws; ws += W_BYTES;
  half_t* WkT = (half_t*)ws; ws += W_BYTES;
  half_t* WvT = (half_t*)ws; ws += W_BYTES;
  half_t* WoT = (half_t*)ws; ws += W_BYTES;
  half_t* Qh  = (half_t*)ws; ws += XH_BYTES;
  half_t* Kh  = (half_t*)ws; ws += XH_BYTES;
  half_t* Vh  = (half_t*)ws; ws += XH_BYTES;
  half_t* Ah  = (half_t*)ws; ws += XH_BYTES;   // ~192 MiB total workspace

  // 1) casts / weight transposes
  {
    int n8 = Mn * Dn / 8;
    cast_f32_to_f16_kernel<<<n8/256, 256, 0, stream>>>(X, Xh, n8);
    dim3 g(Dn/32, Dn/32), blk(32, 8);
    transpose_cast_kernel<<<g, blk, 0, stream>>>(Wq, WqT);
    transpose_cast_kernel<<<g, blk, 0, stream>>>(Wk, WkT);
    transpose_cast_kernel<<<g, blk, 0, stream>>>(Wv, WvT);
    transpose_cast_kernel<<<g, blk, 0, stream>>>(Wo, WoT);
  }
  // 2) Q/K/V projections (Q pre-scaled by 1/sqrt(HD))
  {
    dim3 g(Dn/128, Mn/128);
    const float qscale = 0.08838834764831845f;
    gemm_wmma_kernel<0><<<g, 256, 0, stream>>>(Xh, WqT, Qh, nullptr, nullptr, Mn, Dn, Dn, qscale);
    gemm_wmma_kernel<0><<<g, 256, 0, stream>>>(Xh, WkT, Kh, nullptr, nullptr, Mn, Dn, Dn, 1.0f);
    gemm_wmma_kernel<0><<<g, 256, 0, stream>>>(Xh, WvT, Vh, nullptr, nullptr, Mn, Dn, Dn, 1.0f);
  }
  // 3) local attention: 512 workgroups, 136 KB dynamic LDS each
  {
    const int smem = (128*(HDn+8) + HDn*(128+8) + 16*16*(128+8)) * (int)sizeof(half_t);
    hipFuncSetAttribute((const void*)attn_kernel,
                        hipFuncAttributeMaxDynamicSharedMemorySize, smem);
    attn_kernel<<<Bn*NBn*Hn, 512, smem, stream>>>(Qh, Kh, Vh, Ah);
  }
  // 4) output projection + bias (f32 output)
  {
    dim3 g(Dn/128, Mn/128);
    gemm_wmma_kernel<1><<<g, 256, 0, stream>>>(Ah, WoT, nullptr, out, bo, Mn, Dn, Dn, 1.0f);
  }
}
